// TransformerBlock_50087908606334
// MI455X (gfx1250) — compile-verified, matrix operand order checked by reading, not run on hardware
//
#include <hip/hip_runtime.h>
#include <cstddef>
#include <cstdint>

// ---------------------------------------------------------------------------
// MI455X / gfx1250 transformer block: bf16 WMMA everywhere (f32 accumulate),
// async-to-LDS staging (ASYNCcnt) + ping-pong LDS in the GEMMs.
// ---------------------------------------------------------------------------

typedef __bf16 bf16_t;
typedef __attribute__((ext_vector_type(16))) __bf16 v16bf;
typedef __attribute__((ext_vector_type(8)))  __bf16 v8bf;
typedef __attribute__((ext_vector_type(4)))  __bf16 v4bf;
typedef __attribute__((ext_vector_type(8)))  float  v8f;
typedef __attribute__((ext_vector_type(4)))  float  v4f;

#define D_MODEL 1024
#define N_HEADS 16
#define HEAD_DIM 64
#define D_FF 4096
#define SEQ_T 2048
#define ROWS 4096           // B*T

// ---- workspace layout (bytes) ----
static constexpr size_t OFF_Q    = 0;                       // [B*H, T, 64] bf16   8 MB
static constexpr size_t OFF_K    = OFF_Q    + 8388608;      //                     8 MB
static constexpr size_t OFF_VT   = OFF_K    + 8388608;      // [B*H, 64, T] bf16   8 MB
static constexpr size_t OFF_H1   = OFF_VT   + 8388608;      // [4096,1024]  bf16   8 MB
static constexpr size_t OFF_ATTN = OFF_H1   + 8388608;      // [4096,1024]  bf16   8 MB
static constexpr size_t OFF_X1   = OFF_ATTN + 8388608;      // [4096,1024]  f32   16 MB
static constexpr size_t OFF_H2   = OFF_X1   + 16777216;     // [4096,1024]  bf16   8 MB
static constexpr size_t OFF_FF1  = OFF_H2   + 8388608;      // [4096,4096]  bf16  32 MB
// total: 96 MB

// ---------------------------------------------------------------------------
// helpers
// ---------------------------------------------------------------------------
__device__ __forceinline__ void lds_fence() { asm volatile("" ::: "memory"); }

template<int IMM>
__device__ __forceinline__ float swzx(float x) {
  return __int_as_float(__builtin_amdgcn_ds_swizzle(__float_as_int(x), IMM));
}
// group-of-32 xor swizzles: imm = (xor_mask<<10) | 0x1f
__device__ __forceinline__ float wave_sum(float x) {
  x += swzx<0x401f>(x);   // xor 16
  x += swzx<0x201f>(x);   // xor 8
  x += swzx<0x101f>(x);   // xor 4
  x += swzx<0x081f>(x);   // xor 2
  x += swzx<0x041f>(x);   // xor 1
  return x;
}

__device__ __forceinline__ v8f wmma_bf16(v16bf a, v16bf b, v8f c) {
  return __builtin_amdgcn_wmma_f32_16x16x32_bf16(false, a, false, b, (short)0, c,
                                                 false, false);
}

// CDNA5 async copy global->LDS, 16 bytes per lane (tracked by ASYNCcnt).
__device__ __forceinline__ void async_ld_b128(uint32_t lds_addr, uint64_t gaddr) {
  asm volatile("global_load_async_to_lds_b128 %0, %1, off"
               :: "v"(lds_addr), "v"(gaddr)
               : "memory");
}
__device__ __forceinline__ void wait_async_all() {
  asm volatile("s_wait_asynccnt 0" ::: "memory");
}
__device__ __forceinline__ uint32_t lds_off(const void* p) {
  // generic pointer to __shared__: low 32 bits are the LDS offset
  return (uint32_t)(uintptr_t)p;
}

// Load a 16-bit WMMA A/B fragment row: 32 contiguous bf16 at p cover the
// fragment's K-range.  Per ISA layout, lane half h holds elements
// [h*8 .. h*8+7] and [16+h*8 .. 16+h*8+7]  (two 16-byte groups).
template<typename P>
__device__ __forceinline__ v16bf load_frag(P p, int h) {
  v8bf a = *(const v8bf*)(p + h * 8);
  v8bf b = *(const v8bf*)(p + 16 + h * 8);
  v16bf f;
#pragma unroll
  for (int i = 0; i < 8; ++i) { f[i] = a[i]; f[i + 8] = b[i]; }
  return f;
}

// ---------------------------------------------------------------------------
// LayerNorm + downconvert to bf16.  One block (256 thr, 8 waves) per row.
// ---------------------------------------------------------------------------
__global__ void __launch_bounds__(256)
ln_kernel(const float* __restrict__ x, const float* __restrict__ g,
          const float* __restrict__ b, bf16_t* __restrict__ out) {
  __shared__ float red[2][8];
  const int row = blockIdx.x, tid = threadIdx.x;
  const float4 v = ((const float4*)(x + (size_t)row * D_MODEL))[tid];
  float s = v.x + v.y + v.z + v.w;
  float q = v.x * v.x + v.y * v.y + v.z * v.z + v.w * v.w;
  s = wave_sum(s);
  q = wave_sum(q);
  const int lane = tid & 31, wave = tid >> 5;
  if (lane == 0) { red[0][wave] = s; red[1][wave] = q; }
  __syncthreads();
  float ts = 0.f, tq = 0.f;
#pragma unroll
  for (int i = 0; i < 8; ++i) { ts += red[0][i]; tq += red[1][i]; }
  const float mu = ts * (1.f / D_MODEL);
  const float var = tq * (1.f / D_MODEL) - mu * mu;
  const float rstd = rsqrtf(var + 1e-5f);
  const float4 gv = ((const float4*)g)[tid];
  const float4 bv = ((const float4*)b)[tid];
  v4bf o;
  o[0] = (bf16_t)((v.x - mu) * rstd * gv.x + bv.x);
  o[1] = (bf16_t)((v.y - mu) * rstd * gv.y + bv.y);
  o[2] = (bf16_t)((v.z - mu) * rstd * gv.z + bv.z);
  o[3] = (bf16_t)((v.w - mu) * rstd * gv.w + bv.w);
  *(v4bf*)(out + (size_t)row * D_MODEL + tid * 4) = o;
}

// ---------------------------------------------------------------------------
// Tiled WMMA GEMM:  C[M,N] = A_bf16[M,K] @ W_f32[K,N] + bias, with per-mode
// epilogue.  Block tile 128x128, BK=32, 256 threads = 8 waves, 32x64 / wave.
// A-tile staged with global_load_async_to_lds_b128 into ping-pong LDS.
// ---------------------------------------------------------------------------
enum { GEMM_QKV = 0, GEMM_OUT = 1, GEMM_FF1 = 2, GEMM_FF2 = 3 };

#define LDSA_STRIDE 40   // 32 bf16 + 8 pad = 80 B rows: 16-lane b128 reads hit distinct banks

template<int MODE>
__global__ void __launch_bounds__(256)
gemm_kernel(const bf16_t* __restrict__ A, const float* __restrict__ W,
            const float* __restrict__ bias, const float* __restrict__ resid,
            float* __restrict__ outF, bf16_t* __restrict__ outB,
            bf16_t* __restrict__ qo, bf16_t* __restrict__ ko,
            bf16_t* __restrict__ vto, int N, int K) {
  __shared__ bf16_t lA[2][128 * LDSA_STRIDE];
  __shared__ bf16_t lB[2][128 * LDSA_STRIDE];

  const int tid = threadIdx.x;
  const int bm = blockIdx.y, bn = blockIdx.x;
  const int lane = tid & 31, wave = tid >> 5;
  const int wm = wave >> 1, wn = wave & 1;      // 4 x 2 wave grid
  const int lm = lane & 15, hh = lane >> 4;

  v8f acc[2][4] = {};

  // staging maps
  const int arow = tid & 127, ah = tid >> 7;            // A: 128 rows x 2 halves
  const int bk = tid >> 3, bn0 = (tid & 7) * 16;        // W: 32 k-rows x 8 col groups
  uint64_t gA =
      (uint64_t)(uintptr_t)(A + (size_t)(bm * 128 + arow) * K + ah * 16);
  const float* gW = W + (size_t)bk * N + bn * 128 + bn0;
  const uint32_t laddrA0 = lds_off(&lA[0][arow * LDSA_STRIDE + ah * 16]);
  const uint32_t laddrA1 = lds_off(&lA[1][arow * LDSA_STRIDE + ah * 16]);

  const int ntiles = K >> 5;

  // ---- prologue: stage tile 0 into buffer 0 ----
  async_ld_b128(laddrA0, gA);
  async_ld_b128(laddrA0 + 16, gA + 16);
  gA += 64;
  {
    v4f w0 = *(const v4f*)(gW);
    v4f w1 = *(const v4f*)(gW + 4);
    v4f w2 = *(const v4f*)(gW + 8);
    v4f w3 = *(const v4f*)(gW + 12);
#pragma unroll
    for (int j = 0; j < 4; ++j) {
      lB[0][(bn0 + j) * LDSA_STRIDE + bk]      = (bf16_t)w0[j];
      lB[0][(bn0 + 4 + j) * LDSA_STRIDE + bk]  = (bf16_t)w1[j];
      lB[0][(bn0 + 8 + j) * LDSA_STRIDE + bk]  = (bf16_t)w2[j];
      lB[0][(bn0 + 12 + j) * LDSA_STRIDE + bk] = (bf16_t)w3[j];
    }
    gW += (size_t)32 * N;
  }
  wait_async_all();
  __syncthreads();

  for (int i = 0; i < ntiles; ++i) {
    const int buf = i & 1, nxt = buf ^ 1;
    const bool have_next = (i + 1 < ntiles);
    v4f w0{}, w1{}, w2{}, w3{};
    if (have_next) {
      // async-stage next A tile (overlaps with WMMA below)
      const uint32_t la = (nxt ? laddrA1 : laddrA0);
      async_ld_b128(la, gA);
      async_ld_b128(la + 16, gA + 16);
      gA += 64;
      // issue next W tile loads now; convert after compute
      w0 = *(const v4f*)(gW);
      w1 = *(const v4f*)(gW + 4);
      w2 = *(const v4f*)(gW + 8);
      w3 = *(const v4f*)(gW + 12);
      __builtin_prefetch((const void*)(gW + (size_t)32 * N), 0, 3);
    }

    // --- compute: 2 A-frags x 4 B-frags = 8 WMMAs ---
    const bf16_t* pa = &lA[buf][(wm * 32 + lm) * LDSA_STRIDE];
    v16bf fa0 = load_frag(pa, hh);
    v16bf fa1 = load_frag(pa + 16 * LDSA_STRIDE, hh);
#pragma unroll
    for (int tc = 0; tc < 4; ++tc) {
      v16bf fb =
          load_frag(&lB[buf][(wn * 64 + tc * 16 + lm) * LDSA_STRIDE], hh);
      acc[0][tc] = wmma_bf16(fa0, fb, acc[0][tc]);
      acc[1][tc] = wmma_bf16(fa1, fb, acc[1][tc]);
    }

    if (have_next) {
#pragma unroll
      for (int j = 0; j < 4; ++j) {
        lB[nxt][(bn0 + j) * LDSA_STRIDE + bk]      = (bf16_t)w0[j];
        lB[nxt][(bn0 + 4 + j) * LDSA_STRIDE + bk]  = (bf16_t)w1[j];
        lB[nxt][(bn0 + 8 + j) * LDSA_STRIDE + bk]  = (bf16_t)w2[j];
        lB[nxt][(bn0 + 12 + j) * LDSA_STRIDE + bk] = (bf16_t)w3[j];
      }
      gW += (size_t)32 * N;
    }
    wait_async_all();
    __syncthreads();
  }

  // --- epilogue ---
#pragma unroll
  for (int tr = 0; tr < 2; ++tr) {
#pragma unroll
    for (int tc = 0; tc < 4; ++tc) {
      const int colg = bn * 128 + wn * 64 + tc * 16 + lm;
      const float bv = bias[colg];
#pragma unroll
      for (int r = 0; r < 8; ++r) {
        const int rowg = bm * 128 + wm * 32 + tr * 16 + hh * 8 + r;
        const float val = acc[tr][tc][r] + bv;
        if constexpr (MODE == GEMM_QKV) {
          const int which = colg >> 10;
          const int cc = colg & 1023;
          const int head = cc >> 6, d = cc & 63;
          const int bb = rowg >> 11, t = rowg & 2047;
          const size_t rc = ((size_t)(bb * N_HEADS + head) * SEQ_T + t) * HEAD_DIM + d;
          if (which == 0)       qo[rc] = (bf16_t)(val * 0.125f);   // 1/sqrt(64)
          else if (which == 1)  ko[rc] = (bf16_t)val;
          else vto[((size_t)(bb * N_HEADS + head) * HEAD_DIM + d) * SEQ_T + t] =
                   (bf16_t)val;
        } else if constexpr (MODE == GEMM_FF1) {
          outB[(size_t)rowg * D_FF + colg] = (bf16_t)fmaxf(val, 0.f);
        } else {  // GEMM_OUT / GEMM_FF2: residual add, f32 out
          const size_t idx = (size_t)rowg * D_MODEL + colg;
          outF[idx] = resid[idx] + val;
        }
      }
    }
  }
}

// ---------------------------------------------------------------------------
// Flash attention: one wave = 16 query rows, 32 keys / iteration.
// grid = (T/128, B*H), 256 threads = 8 waves.
// ---------------------------------------------------------------------------
__global__ void __launch_bounds__(256)
attn_kernel(const bf16_t* __restrict__ Qm, const bf16_t* __restrict__ Km,
            const bf16_t* __restrict__ VTm, bf16_t* __restrict__ out) {
  __shared__ float sS[8][16 * 36];   // per-wave S tile, stride 36 (16B aligned, conflict-free)
  __shared__ float sBc[8][16];       // per-wave row-broadcast scratch

  const int tid = threadIdx.x, lane = tid & 31, wave = tid >> 5;
  const int lm = lane & 15, hh = lane >> 4;
  const int bh = blockIdx.y;
  const int qbase = blockIdx.x * 128 + wave * 16;

  const bf16_t* q  = Qm  + (size_t)bh * SEQ_T * HEAD_DIM;
  const bf16_t* k  = Km  + (size_t)bh * SEQ_T * HEAD_DIM;
  const bf16_t* vt = VTm + (size_t)bh * HEAD_DIM * SEQ_T;

  // resident Q fragments (16x64 over two 32-wide d-chunks); Q is pre-scaled.
  const v16bf fq0 = load_frag(q + (size_t)(qbase + lm) * HEAD_DIM, hh);
  const v16bf fq1 = load_frag(q + (size_t)(qbase + lm) * HEAD_DIM + 32, hh);

  v8f O[4] = {};
  float mrow = -__builtin_inff(), lrow = 0.f;
  float* S  = sS[wave];
  float* Bc = sBc[wave];
  const int qg = qbase + lm;
  const int qmax = qbase + 15;

  for (int c0 = 0; c0 <= qmax; c0 += 32) {
    // S = Q K^T for 32 keys: two 16x16 tiles, 2 WMMAs each over d
    v8f ca = {}, cb = {};
    {
      v16bf f0 = load_frag(k + (size_t)(c0 + lm) * HEAD_DIM, hh);
      v16bf f1 = load_frag(k + (size_t)(c0 + lm) * HEAD_DIM + 32, hh);
      ca = wmma_bf16(fq0, f0, ca);
      ca = wmma_bf16(fq1, f1, ca);
    }
    {
      v16bf f0 = load_frag(k + (size_t)(c0 + 16 + lm) * HEAD_DIM, hh);
      v16bf f1 = load_frag(k + (size_t)(c0 + 16 + lm) * HEAD_DIM + 32, hh);
      cb = wmma_bf16(fq0, f0, cb);
      cb = wmma_bf16(fq1, f1, cb);
    }
    // issue V-fragment loads early: their latency hides behind the softmax
    v16bf fv[4];
#pragma unroll
    for (int f = 0; f < 4; ++f)
      fv[f] = load_frag(vt + (size_t)(f * 16 + lm) * SEQ_T + c0, hh);
    // prefetch next chunk's K rows
    if (c0 + 32 <= qmax) {
      __builtin_prefetch((const void*)(k + (size_t)(c0 + 32 + lm) * HEAD_DIM), 0, 3);
      __builtin_prefetch((const void*)(k + (size_t)(c0 + 48 + lm) * HEAD_DIM), 0, 3);
    }
    // spill S tile (C layout) to per-wave LDS; same-wave LDS ops are in-order.
#pragma unroll
    for (int r = 0; r < 8; ++r) {
      S[(hh * 8 + r) * 36 + lm]      = ca[r];
      S[(hh * 8 + r) * 36 + 16 + lm] = cb[r];
    }
    lds_fence();
    // chunk row-max (causal masked); lane owns row lm, half hh covers 16 keys
    float cmax = -__builtin_inff();
#pragma unroll
    for (int j = 0; j < 16; ++j) {
      const int kg = c0 + hh * 16 + j;
      const float s = S[lm * 36 + hh * 16 + j];
      cmax = (kg <= qg && s > cmax) ? s : cmax;
    }
    cmax = fmaxf(cmax, swzx<0x401f>(cmax));
    const float mnew = fmaxf(mrow, cmax);
    const float alpha = __expf(mrow - mnew);
    lrow *= alpha;
    // build P as bf16 A-fragment (kk layout) + chunk row-sum
    v16bf pf;
    float rs = 0.f;
#pragma unroll
    for (int i = 0; i < 8; ++i) {
      const int kj = hh * 8 + i;
      const float s = S[lm * 36 + kj];
      const float p = (c0 + kj <= qg) ? __expf(s - mnew) : 0.f;
      rs += p;
      pf[i] = (bf16_t)p;
    }
#pragma unroll
    for (int i = 0; i < 8; ++i) {
      const int kj = 16 + hh * 8 + i;
      const float s = S[lm * 36 + kj];
      const float p = (c0 + kj <= qg) ? __expf(s - mnew) : 0.f;
      rs += p;
      pf[8 + i] = (bf16_t)p;
    }
    rs += swzx<0x401f>(rs);   // combine the two half-wave partial sums
    lrow += rs;
    mrow = mnew;
    // broadcast alpha per row into C layout and rescale O
    if (hh == 0) Bc[lm] = alpha;
    lds_fence();
    float ar[8];
#pragma unroll
    for (int r = 0; r < 8; ++r) ar[r] = Bc[hh * 8 + r];
    lds_fence();
#pragma unroll
    for (int f = 0; f < 4; ++f) {
#pragma unroll
      for (int r = 0; r < 8; ++r) O[f][r] *= ar[r];
      O[f] = wmma_bf16(pf, fv[f], O[f]);
    }
  }

  // normalize and store (concat-heads bf16 layout [B*T, 1024])
  if (hh == 0) Bc[lm] = 1.0f / lrow;
  lds_fence();
  float ir[8];
#pragma unroll
  for (int r = 0; r < 8; ++r) ir[r] = Bc[hh * 8 + r];
  const int bb = bh >> 4, head = bh & 15;
#pragma unroll
  for (int f = 0; f < 4; ++f) {
#pragma unroll
    for (int r = 0; r < 8; ++r) {
      const int t = qbase + hh * 8 + r;
      out[((size_t)(bb * SEQ_T + t)) * D_MODEL + head * HEAD_DIM + f * 16 + lm] =
          (bf16_t)(O[f][r] * ir[r]);
    }
  }
}

// ---------------------------------------------------------------------------
extern "C" void kernel_launch(void* const* d_in, const int* in_sizes, int n_in,
                              void* d_out, int out_size, void* d_ws, size_t ws_size,
                              hipStream_t stream) {
  (void)in_sizes; (void)n_in; (void)out_size; (void)ws_size;
  const float* x     = (const float*)d_in[0];
  const float* ln1_g = (const float*)d_in[1];
  const float* ln1_b = (const float*)d_in[2];
  const float* w_qkv = (const float*)d_in[3];
  const float* b_qkv = (const float*)d_in[4];
  const float* w_out = (const float*)d_in[5];
  const float* b_out = (const float*)d_in[6];
  const float* ln2_g = (const float*)d_in[7];
  const float* ln2_b = (const float*)d_in[8];
  const float* w_ff1 = (const float*)d_in[9];
  const float* b_ff1 = (const float*)d_in[10];
  const float* w_ff2 = (const float*)d_in[11];
  const float* b_ff2 = (const float*)d_in[12];
  float* outp = (float*)d_out;

  char* ws = (char*)d_ws;
  bf16_t* qb    = (bf16_t*)(ws + OFF_Q);
  bf16_t* kb    = (bf16_t*)(ws + OFF_K);
  bf16_t* vtb   = (bf16_t*)(ws + OFF_VT);
  bf16_t* h1    = (bf16_t*)(ws + OFF_H1);
  bf16_t* attnb = (bf16_t*)(ws + OFF_ATTN);
  float*  x1    = (float*) (ws + OFF_X1);
  bf16_t* h2    = (bf16_t*)(ws + OFF_H2);
  bf16_t* ff1   = (bf16_t*)(ws + OFF_FF1);

  // 1) LN1 -> bf16
  ln_kernel<<<ROWS, 256, 0, stream>>>(x, ln1_g, ln1_b, h1);
  // 2) QKV projection, scatter Q/K/V^T
  gemm_kernel<GEMM_QKV><<<dim3(3 * D_MODEL / 128, ROWS / 128), 256, 0, stream>>>(
      h1, w_qkv, b_qkv, nullptr, nullptr, nullptr, qb, kb, vtb, 3 * D_MODEL, D_MODEL);
  // 3) causal flash attention
  attn_kernel<<<dim3(SEQ_T / 128, 2 * N_HEADS), 256, 0, stream>>>(qb, kb, vtb, attnb);
  // 4) output projection + residual -> x1 (f32)
  gemm_kernel<GEMM_OUT><<<dim3(D_MODEL / 128, ROWS / 128), 256, 0, stream>>>(
      attnb, w_out, b_out, x, x1, nullptr, nullptr, nullptr, nullptr, D_MODEL, D_MODEL);
  // 5) LN2 -> bf16
  ln_kernel<<<ROWS, 256, 0, stream>>>(x1, ln2_g, ln2_b, h2);
  // 6) FF1 + ReLU -> bf16
  gemm_kernel<GEMM_FF1><<<dim3(D_FF / 128, ROWS / 128), 256, 0, stream>>>(
      h2, w_ff1, b_ff1, nullptr, nullptr, ff1, nullptr, nullptr, nullptr, D_FF, D_MODEL);
  // 7) FF2 + residual -> out (f32)
  gemm_kernel<GEMM_FF2><<<dim3(D_MODEL / 128, ROWS / 128), 256, 0, stream>>>(
      ff1, w_ff2, b_ff2, x1, outp, nullptr, nullptr, nullptr, nullptr, D_MODEL, D_FF);
}
